// Attention_57423712748130
// MI455X (gfx1250) — matrix operand order repaired and checked by reading, not compiled, not run on hardware
//
#include <hip/hip_runtime.h>
#include <stdint.h>

// ---------------------------------------------------------------------------
// MI455X / gfx1250 implementation.  bf16 WMMA (16x16x32) everywhere, f32 acc.
// Async-to-LDS staging (ASYNCcnt) for the shared K tile in attention.
// ---------------------------------------------------------------------------

typedef __attribute__((ext_vector_type(16))) __bf16 v16bf;
typedef __attribute__((ext_vector_type(8)))  float  v8f;

constexpr int Bc  = 4;      // batch
constexpr int Cc  = 256;    // channels
constexpr int Nc  = 4096;   // H*W
constexpr int CKV = 128;    // kv conv out channels
constexpr int HD  = 64;     // head dim
constexpr int Mkv = 1024;   // kv sequence length

__device__ __forceinline__ uint16_t f2bf(float f) {
    uint32_t u = __float_as_uint(f);
    u += 0x7FFFu + ((u >> 16) & 1u);   // round-to-nearest-even
    return (uint16_t)(u >> 16);
}

// A-matrix fragment, 16x32 bf16, from row-major matrix (row stride ld elems).
// ISA layout: lanes 0-15 row M=lane hold K=0-7 (v0..v3) and K=16-23 (v4..v7);
// lanes 16-31 same rows hold K=8-15 and K=24-31.  K-pairs are dword reads.
__device__ __forceinline__ v16bf load_a_frag(const uint16_t* base, int ld, int lane) {
    const uint32_t* row = (const uint32_t*)(base + (size_t)(lane & 15) * ld);
    int kb2 = (lane >> 4) * 4;   // kbase/2
    union { uint32_t u[8]; v16bf v; } f;
#pragma unroll
    for (int j = 0; j < 4; ++j) { f.u[j] = row[kb2 + j]; f.u[4 + j] = row[kb2 + 8 + j]; }
    return f.v;
}

// B-matrix fragment, 32x16 bf16, where B[k][n] = rowmajor[n][k] (ld = row stride).
// ISA layout: col = lane&15; lanes 0-15 hold K=0-7 and K=16-23, lanes 16-31 hold
// K=8-15 and K=24-31, two K per dword.
__device__ __forceinline__ v16bf load_bt_frag(const uint16_t* base, int ld, int lane) {
    const uint32_t* row = (const uint32_t*)(base + (size_t)(lane & 15) * ld);
    int kb2 = (lane >> 4) * 4;   // kbase/2, kbase = (lane>>4)*8
    union { uint32_t u[8]; v16bf v; } f;
#pragma unroll
    for (int j = 0; j < 4; ++j) { f.u[j] = row[kb2 + j]; f.u[4 + j] = row[kb2 + 8 + j]; }
    return f.v;
}

__device__ __forceinline__ v8f wmma_bf16(v16bf a, v16bf b, v8f c) {
    return __builtin_amdgcn_wmma_f32_16x16x32_bf16(false, a, false, b, (short)0, c,
                                                   false, false);
}

// ---------------------------------------------------------------------------
// Kernel 1: fold BN constants, convert weights f32 -> bf16.
// ---------------------------------------------------------------------------
__global__ void setup_kernel(const float* __restrict__ qw, const float* __restrict__ kvw,
                             const float* __restrict__ pw,
                             const float* __restrict__ qg, const float* __restrict__ qb,
                             const float* __restrict__ qm, const float* __restrict__ qv,
                             const float* __restrict__ kg, const float* __restrict__ kb,
                             const float* __restrict__ km, const float* __restrict__ kv,
                             uint16_t* __restrict__ wq, uint16_t* __restrict__ wkv,
                             uint16_t* __restrict__ wp,
                             float* __restrict__ gq, float* __restrict__ bbq,
                             float* __restrict__ gkv, float* __restrict__ bbkv) {
    int i = blockIdx.x * blockDim.x + threadIdx.x;
    if (i < 65536)       wq[i]          = f2bf(qw[i]);
    else if (i < 98304)  wkv[i - 65536] = f2bf(kvw[i - 65536]);
    else if (i < 163840) wp[i - 98304]  = f2bf(pw[i - 98304]);
    else if (i < 164096) {
        int o = i - 163840;
        float g = qg[o] * rsqrtf(qv[o] + 1e-5f);
        gq[o] = g; bbq[o] = qb[o] - qm[o] * g;
    } else if (i < 164224) {
        int o = i - 164096;
        float g = kg[o] * rsqrtf(kv[o] + 1e-5f);
        gkv[o] = g; bbkv[o] = kb[o] - km[o] * g;
    }
}

// ---------------------------------------------------------------------------
// Kernel 2: transpose (C,N) f32 -> (N,C) bf16 per batch (so conv B-fragments
// have K=c contiguous).  32x32 tiles through LDS, coalesced both ways.
// ---------------------------------------------------------------------------
__global__ void transpose_cvt(const float* __restrict__ in, uint16_t* __restrict__ out) {
    __shared__ float t[32][33];
    int b  = blockIdx.z;
    int p0 = blockIdx.x * 32, c0 = blockIdx.y * 32;
    const float* ib = in  + (size_t)b * Cc * Nc;
    uint16_t*    ob = out + (size_t)b * Cc * Nc;
    int tx = threadIdx.x, ty = threadIdx.y;   // 32 x 8
#pragma unroll
    for (int i = 0; i < 4; ++i)
        t[ty + 8 * i][tx] = ib[(size_t)(c0 + ty + 8 * i) * Nc + p0 + tx];
    __syncthreads();
#pragma unroll
    for (int i = 0; i < 4; ++i)
        ob[(size_t)(p0 + ty + 8 * i) * Cc + c0 + tx] = f2bf(t[tx][ty + 8 * i]);
}

// ---------------------------------------------------------------------------
// Kernel 3: 1x1 conv (GEMM  W[Co,256] @ X[256,N]) + BN + SiLU, bf16 out in
// flat (Co,N) per batch.  Wave computes 32(o) x 64(p): each B fragment feeds
// two WMMAs, each A fragment feeds four.  Block = 4 waves (split over p).
// ---------------------------------------------------------------------------
__global__ void conv_bn_silu(const uint16_t* __restrict__ Xt, const uint16_t* __restrict__ W,
                             const float* __restrict__ g, const float* __restrict__ bb,
                             uint16_t* __restrict__ Out, int Cout) {
    int lane = threadIdx.x & 31, wave = threadIdx.x >> 5;
    int b  = blockIdx.x;
    int o0 = blockIdx.y * 32;
    int p0 = blockIdx.z * 256 + wave * 64;
    const uint16_t* Xb = Xt  + (size_t)b * Cc * Nc;
    uint16_t*       Ob = Out + (size_t)b * Cout * Nc;

    v8f acc[8] = {};
    for (int c0 = 0; c0 < Cc; c0 += 32) {
        v16bf a0 = load_a_frag(W + (size_t)o0 * Cc + c0,        Cc, lane);
        v16bf a1 = load_a_frag(W + (size_t)(o0 + 16) * Cc + c0, Cc, lane);
#pragma unroll
        for (int t = 0; t < 4; ++t) {
            v16bf bf = load_bt_frag(Xb + (size_t)(p0 + t * 16) * Cc + c0, Cc, lane);
            acc[t]     = wmma_bf16(a0, bf, acc[t]);
            acc[4 + t] = wmma_bf16(a1, bf, acc[4 + t]);
        }
    }
    int rbase = (lane >> 4) * 8, col = lane & 15;
#pragma unroll
    for (int r = 0; r < 2; ++r)
#pragma unroll
        for (int t = 0; t < 4; ++t) {
            int p = p0 + t * 16 + col;
#pragma unroll
            for (int j = 0; j < 8; ++j) {
                int o = o0 + r * 16 + rbase + j;
                float v = acc[r * 4 + t][j] * g[o] + bb[o];
                float s = v / (1.0f + __expf(-v));           // SiLU
                Ob[(size_t)o * Nc + p] = f2bf(s);
            }
        }
}

// ---------------------------------------------------------------------------
// Kernel 4: flash attention per (b,h).  Wave owns 16 q rows x full hd=64.
// K tile (shared by all 4 waves) is staged into LDS once per workgroup via
// GLOBAL_LOAD_ASYNC_TO_LDS_B128 (ASYNCcnt), V tile is transposed into LDS,
// P goes through LDS to convert C-layout -> A-layout.  Next kv tile is
// prefetched with global_prefetch_b8.
// ---------------------------------------------------------------------------
__global__ void attn_kernel(const uint16_t* __restrict__ Q, const uint16_t* __restrict__ KV,
                            uint16_t* __restrict__ O) {
    constexpr int KST = 72;                 // K tile LDS row stride (16B aligned, padded)
    __shared__ uint16_t Ks[64 * KST];       // K tile, row-major (m, d)
    __shared__ uint16_t Vs[64 * 66];        // V^T tile: Vs[d][m], stride 66
    __shared__ uint16_t Ps[4][16 * 64];     // per-wave P tile, row stride 64
    int lane = threadIdx.x & 31, wave = threadIdx.x >> 5;
    int bh = blockIdx.y; int b = bh >> 2, h = bh & 3;
    int n0 = blockIdx.x * 64 + wave * 16;
    const uint16_t* Qb  = Q  + (size_t)b * Cc * Nc;
    const uint16_t* KVb = KV + (size_t)b * CKV * Nc;
    uint16_t*       Ob  = O  + (size_t)b * Cc * Nc + (size_t)h * Nc * HD;

    v16bf aq[2];
    aq[0] = load_a_frag(Qb + (size_t)n0 * Cc + h * HD,      Cc, lane);
    aq[1] = load_a_frag(Qb + (size_t)n0 * Cc + h * HD + 32, Cc, lane);

    v8f acco[4] = {};
    float mrun[8], lrun[8];
#pragma unroll
    for (int j = 0; j < 8; ++j) { mrun[j] = -1e30f; lrun[j] = 0.0f; }
    const float SC = 0.125f;   // hd^-0.5

    for (int m0 = 0; m0 < Mkv; m0 += 64) {
        __syncthreads();   // waits DS ops; prior-tile LDS reads are done

        // --- async-stage K tile (64 rows x 128B) straight into LDS ----------
        {
            int t = threadIdx.x;
#pragma unroll
            for (int i = 0; i < 4; ++i) {
                int chunk = i * 128 + t;           // 512 x b128 chunks
                int row = chunk >> 3, seg = chunk & 7;
                uint64_t gaddr = (uint64_t)(uintptr_t)
                    (KVb + (size_t)(m0 + row) * 512 + h * HD + seg * 8);
                uint32_t laddr = (uint32_t)(uintptr_t)(&Ks[row * KST + seg * 8]);
                asm volatile("global_load_async_to_lds_b128 %0, %1, off"
                             :: "v"(laddr), "v"(gaddr) : "memory");
            }
        }
        // --- stage V tile transposed (needs element transpose -> VGPR path) -
        {
            int d  = threadIdx.x & 63;
            int r0 = threadIdx.x >> 6;
#pragma unroll
            for (int i = 0; i < 32; ++i) {
                int m = i * 2 + r0;
                Vs[d * 66 + m] = KVb[(size_t)(m0 + m) * 512 + Cc + h * HD + d];
            }
        }
        // --- prefetch next kv tile (one K or V row per thread) ---------------
        if (m0 + 64 < Mkv) {
            int row = threadIdx.x & 63, half = threadIdx.x >> 6;
            __builtin_prefetch(KVb + (size_t)(m0 + 64 + row) * 512 + half * Cc + h * HD,
                               0, 1);
        }
        asm volatile("s_wait_asynccnt 0x0" ::: "memory");
        __syncthreads();

        // --- S = q k^T  (16 x 64 tile, K = hd = 64), K frags from LDS -------
        v8f s[4] = {};
#pragma unroll
        for (int t = 0; t < 4; ++t)
#pragma unroll
            for (int ks = 0; ks < 2; ++ks) {
                v16bf bk = load_bt_frag(&Ks[(t * 16) * KST + ks * 32], KST, lane);
                s[t] = wmma_bf16(aq[ks], bk, s[t]);
            }

        // --- online softmax (row stats replicated across 16-lane halves) ---
        float p[4][8];
#pragma unroll
        for (int j = 0; j < 8; ++j) {
            float mt = fmaxf(fmaxf(s[0][j], s[1][j]), fmaxf(s[2][j], s[3][j])) * SC;
#pragma unroll
            for (int msk = 1; msk < 16; msk <<= 1) mt = fmaxf(mt, __shfl_xor(mt, msk, 32));
            float mnew  = fmaxf(mrun[j], mt);
            float alpha = __expf(mrun[j] - mnew);
            float rs = 0.f;
#pragma unroll
            for (int t = 0; t < 4; ++t) {
                float e = __expf(s[t][j] * SC - mnew);
                p[t][j] = e; rs += e;
            }
#pragma unroll
            for (int msk = 1; msk < 16; msk <<= 1) rs += __shfl_xor(rs, msk, 32);
            lrun[j] = lrun[j] * alpha + rs;
            mrun[j] = mnew;
#pragma unroll
            for (int t = 0; t < 4; ++t) acco[t][j] *= alpha;
        }

        {   // stage P (C-layout -> row-major LDS tile)
            int rbase = (lane >> 4) * 8, col = lane & 15;
#pragma unroll
            for (int t = 0; t < 4; ++t)
#pragma unroll
                for (int j = 0; j < 8; ++j)
                    Ps[wave][(rbase + j) * 64 + t * 16 + col] = f2bf(p[t][j]);
        }
        __syncthreads();

        // --- O += P V  (K over the 64 kv of this tile) ----------------------
#pragma unroll
        for (int ks = 0; ks < 2; ++ks) {
            v16bf ap = load_a_frag(&Ps[wave][ks * 32], 64, lane);
#pragma unroll
            for (int dt = 0; dt < 4; ++dt) {
                v16bf bv = load_bt_frag(&Vs[(dt * 16) * 66 + ks * 32], 66, lane);
                acco[dt] = wmma_bf16(ap, bv, acco[dt]);
            }
        }
    }

    int rbase = (lane >> 4) * 8, col = lane & 15;
#pragma unroll
    for (int dt = 0; dt < 4; ++dt)
#pragma unroll
        for (int j = 0; j < 8; ++j) {
            float v = acco[dt][j] / lrun[j];
            Ob[(size_t)(n0 + rbase + j) * HD + dt * 16 + col] = f2bf(v);
        }
}

// ---------------------------------------------------------------------------
// Kernel 5: projection GEMM out[n,co] = sum_c O[n,c] * Wp[co,c] + pb[co], f32.
// Wave computes 32(n) x 64(co) for fragment reuse.
// ---------------------------------------------------------------------------
__global__ void proj_kernel(const uint16_t* __restrict__ O, const uint16_t* __restrict__ Wp,
                            const float* __restrict__ pb, float* __restrict__ out) {
    int lane = threadIdx.x & 31, wave = threadIdx.x >> 5;
    int b   = blockIdx.z;
    int n0  = blockIdx.x * 128 + wave * 32;
    int co0 = blockIdx.y * 64;
    const uint16_t* Ob = O + (size_t)b * Cc * Nc;
    float* outb = out + (size_t)b * Nc * Cc;

    v8f acc[8] = {};
    for (int c0 = 0; c0 < Cc; c0 += 32) {
        v16bf a0 = load_a_frag(Ob + (size_t)n0 * Cc + c0,        Cc, lane);
        v16bf a1 = load_a_frag(Ob + (size_t)(n0 + 16) * Cc + c0, Cc, lane);
#pragma unroll
        for (int t = 0; t < 4; ++t) {
            v16bf bf = load_bt_frag(Wp + (size_t)(co0 + t * 16) * Cc + c0, Cc, lane);
            acc[t]     = wmma_bf16(a0, bf, acc[t]);
            acc[4 + t] = wmma_bf16(a1, bf, acc[4 + t]);
        }
    }
    int rbase = (lane >> 4) * 8, col = lane & 15;
#pragma unroll
    for (int r = 0; r < 2; ++r)
#pragma unroll
        for (int t = 0; t < 4; ++t) {
            int co = co0 + t * 16 + col;
            float bias = pb[co];
#pragma unroll
            for (int j = 0; j < 8; ++j)
                outb[(size_t)(n0 + r * 16 + rbase + j) * Cc + co] = acc[r * 4 + t][j] + bias;
        }
}

// ---------------------------------------------------------------------------
extern "C" void kernel_launch(void* const* d_in, const int* in_sizes, int n_in,
                              void* d_out, int out_size, void* d_ws, size_t ws_size,
                              hipStream_t stream) {
    (void)in_sizes; (void)n_in; (void)out_size; (void)ws_size;
    const float* x    = (const float*)d_in[0];
    const float* y    = (const float*)d_in[1];
    const float* q_w  = (const float*)d_in[4];
    const float* q_g  = (const float*)d_in[5];
    const float* q_b  = (const float*)d_in[6];
    const float* q_m  = (const float*)d_in[7];
    const float* q_v  = (const float*)d_in[8];
    const float* kv_w = (const float*)d_in[9];
    const float* kv_g = (const float*)d_in[10];
    const float* kv_b = (const float*)d_in[11];
    const float* kv_m = (const float*)d_in[12];
    const float* kv_v = (const float*)d_in[13];
    const float* p_w  = (const float*)d_in[14];
    const float* p_b  = (const float*)d_in[15];

    char* ws = (char*)d_ws;
    size_t off = 0;
    auto alloc = [&](size_t bytes) -> void* {
        void* p = ws + off;
        off += (bytes + 255) & ~(size_t)255;
        return p;
    };
    uint16_t* Xt    = (uint16_t*)alloc(sizeof(uint16_t) * Bc * Cc * Nc);
    uint16_t* Yt    = (uint16_t*)alloc(sizeof(uint16_t) * Bc * Cc * Nc);
    uint16_t* Wq    = (uint16_t*)alloc(sizeof(uint16_t) * Cc * Cc);
    uint16_t* Wkv   = (uint16_t*)alloc(sizeof(uint16_t) * CKV * Cc);
    uint16_t* Wp    = (uint16_t*)alloc(sizeof(uint16_t) * Cc * Cc);
    float*    gq    = (float*)alloc(sizeof(float) * Cc);
    float*    bbq   = (float*)alloc(sizeof(float) * Cc);
    float*    gkv   = (float*)alloc(sizeof(float) * CKV);
    float*    bbkv  = (float*)alloc(sizeof(float) * CKV);
    uint16_t* Qact  = (uint16_t*)alloc(sizeof(uint16_t) * Bc * Cc * Nc);
    uint16_t* KVact = (uint16_t*)alloc(sizeof(uint16_t) * Bc * CKV * Nc);
    uint16_t* Oact  = (uint16_t*)alloc(sizeof(uint16_t) * Bc * Cc * Nc);

    setup_kernel<<<dim3((164224 + 255) / 256), 256, 0, stream>>>(
        q_w, kv_w, p_w, q_g, q_b, q_m, q_v, kv_g, kv_b, kv_m, kv_v,
        Wq, Wkv, Wp, gq, bbq, gkv, bbkv);

    transpose_cvt<<<dim3(Nc / 32, Cc / 32, Bc), dim3(32, 8), 0, stream>>>(x, Xt);
    transpose_cvt<<<dim3(Nc / 32, Cc / 32, Bc), dim3(32, 8), 0, stream>>>(y, Yt);

    conv_bn_silu<<<dim3(Bc, Cc / 32,  Nc / 256), 128, 0, stream>>>(Xt, Wq,  gq,  bbq,  Qact,  Cc);
    conv_bn_silu<<<dim3(Bc, CKV / 32, Nc / 256), 128, 0, stream>>>(Yt, Wkv, gkv, bbkv, KVact, CKV);

    attn_kernel<<<dim3(Nc / 64, Bc * 4), 128, 0, stream>>>(Qact, KVact, Oact);

    proj_kernel<<<dim3(Nc / 128, Cc / 64, Bc), 128, 0, stream>>>(Oact, Wp, p_b, (float*)d_out);
}
